// STEncoderBlock_50096498540657
// MI455X (gfx1250) — matrix operand order repaired and checked by reading, not compiled, not run on hardware
//
#include <hip/hip_runtime.h>
#include <hip/hip_bf16.h>

// ---------------- problem constants ----------------
#define B_    8
#define T_    12
#define N_    325
#define BT_   (B_ * T_)
#define Dd    64
#define Mreal 975          // 3*N
#define MP    992          // padded M (31 pair-tiles of 32, 62 row tiles of 16)
#define NR    336          // N padded to 16  (row tiles for gso GEMMs)
#define NP    352          // N padded to 32  (K tiles for gso GEMMs)
#define NEGBIG (-1e30f)

typedef __attribute__((ext_vector_type(16))) _Float16 v16h;
typedef __attribute__((ext_vector_type(8)))  float    v8f;

#define V8Z {0.f,0.f,0.f,0.f,0.f,0.f,0.f,0.f}

// ---------------- WMMA helpers (gfx1250, wave32) ----------------
__device__ __forceinline__ v8f wmma16(v16h a, v16h b, v8f c) {
  // D = A(16x32 f16) * B(32x16 f16) + C(16x16 f32)
  return __builtin_amdgcn_wmma_f32_16x16x32_f16(false, a, false, b, (short)0, c, false, false);
}

// A fragment: lane holds row (lane&15); element i <-> K = kg*8+i, element 8+i <-> K = 16+kg*8+i
__device__ __forceinline__ v16h a_frag_h(const _Float16* rowptr, int lane) {
  int kg = (lane >> 4) & 1;
  const _Float16* p = rowptr + kg * 8;
  v16h a;
#pragma unroll
  for (int i = 0; i < 8; ++i) { a[i] = p[i]; a[i + 8] = p[i + 16]; }
  return a;
}
__device__ __forceinline__ v16h a_frag_f(const float* rowptr, int lane, bool valid) {
  int kg = (lane >> 4) & 1;
  const float* p = rowptr + kg * 8;
  v16h a;
#pragma unroll
  for (int i = 0; i < 8; ++i) {
    a[i]     = valid ? (_Float16)p[i]      : (_Float16)0.f;
    a[i + 8] = valid ? (_Float16)p[i + 16] : (_Float16)0.f;
  }
  return a;
}
// B fragment: lane holds column (lane&15); element j <-> K = kg*8+j, element 8+j <-> K = 16+kg*8+j.
// colptr must point at K=0 of this lane's column laid out contiguously in K (i.e. B^T row).
__device__ __forceinline__ v16h b_frag_h(const _Float16* colptr, int lane) {
  int kg = (lane >> 4) & 1;
  const _Float16* p = colptr + kg * 8;
  v16h b;
#pragma unroll
  for (int i = 0; i < 8; ++i) { b[i] = p[i]; b[i + 8] = p[i + 16]; }
  return b;
}
__device__ __forceinline__ v16h b_frag_f(const float* colptr, int lane) {
  int kg = (lane >> 4) & 1;
  const float* p = colptr + kg * 8;
  v16h b;
#pragma unroll
  for (int i = 0; i < 8; ++i) { b[i] = (_Float16)p[i]; b[i + 8] = (_Float16)p[i + 16]; }
  return b;
}

__device__ __forceinline__ float bn_eval(float y, float g, float be) {
  return y * (g * rsqrtf(1.f + 1e-5f)) + be;
}

// ---------------- prep kernels ----------------
__global__ void k_prep_gso(const float* __restrict__ gso, _Float16* __restrict__ gsoh) {
  size_t idx = (size_t)blockIdx.x * 256 + threadIdx.x;
  if (idx >= (size_t)B_ * NR * NP) return;
  int b = idx / (NR * NP);
  int r = (idx / NP) % NR;
  int c = idx % NP;
  float v = (r < N_ && c < N_) ? gso[((size_t)b * N_ + r) * N_ + c] : 0.f;
  gsoh[idx] = (_Float16)v;
}

__global__ void k_prep_cheb(const float* __restrict__ cheb, _Float16* __restrict__ chebT) {
  int idx = blockIdx.x * 256 + threadIdx.x;
  if (idx >= 3 * 64 * 64) return;
  int k = idx / 4096;
  int j = (idx / 64) % 64;
  int i = idx % 64;
  chebT[idx] = (_Float16)cheb[((size_t)k * 64 + i) * 64 + j];  // chebT[k][j][i] = W_k[i][j]
}

// X0t[bt][w][d][i] = stX[b,t, w*N+i, d] (f16, transposed per window, zero padded to NP)
__global__ void k_prep_x0t(const float* __restrict__ x, _Float16* __restrict__ X0t) {
  size_t idx = (size_t)blockIdx.x * 256 + threadIdx.x;
  if (idx >= (size_t)BT_ * 3 * 64 * NP) return;
  int i  = idx % NP;
  int d  = (idx / NP) % 64;
  int w  = (idx / ((size_t)NP * 64)) % 3;
  int bt = idx / ((size_t)NP * 64 * 3);
  int b = bt / T_, t = bt % T_;
  int tc = t + w - 1; tc = tc < 0 ? 0 : (tc >= T_ ? T_ - 1 : tc);
  float v = (i < N_) ? x[(((size_t)b * T_ + tc) * N_ + i) * 64 + d] : 0.f;
  X0t[idx] = (_Float16)v;
}

// ---------------- QKV projection (WMMA) ----------------
// block = 96 threads (3 waves: Q, K, V). grid = (BT, MP/16)
__global__ __launch_bounds__(96) void k_qkv(const float* __restrict__ x,
                                            const float* __restrict__ Wq,
                                            const float* __restrict__ Wk,
                                            const float* __restrict__ Wv,
                                            _Float16* __restrict__ Qh,
                                            _Float16* __restrict__ Kh,
                                            _Float16* __restrict__ Vt) {
  int bt = blockIdx.x, tile = blockIdx.y;
  int wave = threadIdx.x >> 5, lane = threadIdx.x & 31;
  int ln = lane & 15, kg = (lane >> 4) & 1;
  int b = bt / T_, t = bt % T_;
  int m0 = tile * 16;
  int m = m0 + ln;
  bool valid = m < Mreal;
  const float* rowp = x;
  if (valid) {
    int w = m / N_;
    int n = m - w * N_;
    int tc = t + w - 1; tc = tc < 0 ? 0 : (tc >= T_ ? T_ - 1 : tc);
    rowp = x + (((size_t)b * T_ + tc) * N_ + n) * 64;
  }
  const float* W = (wave == 0) ? Wq : ((wave == 1) ? Wk : Wv);
  v16h a0 = a_frag_f(rowp,      lane, valid);
  v16h a1 = a_frag_f(rowp + 32, lane, valid);
  v8f c[4] = {V8Z, V8Z, V8Z, V8Z};
#pragma unroll
  for (int ot = 0; ot < 4; ++ot) {
    // q[m,o] = sum_d X[m,d]*W[o,d]  ->  B^T row o = W row o (contiguous in d)
    const float* wrow = W + (size_t)(ot * 16 + ln) * 64;
    c[ot] = wmma16(a0, b_frag_f(wrow,      lane), c[ot]);
    c[ot] = wmma16(a1, b_frag_f(wrow + 32, lane), c[ot]);
  }
#pragma unroll
  for (int ot = 0; ot < 4; ++ot)
#pragma unroll
    for (int r = 0; r < 8; ++r) {
      int row = m0 + r + kg * 8;
      int o = ot * 16 + ln;
      if (wave == 2) Vt[((size_t)bt * 64 + o) * MP + row] = (_Float16)c[ot][r];
      else {
        _Float16* dst = (wave == 0) ? Qh : Kh;
        dst[((size_t)bt * MP + row) * 64 + o] = (_Float16)c[ot][r];
      }
    }
}

// ---------------- flash attention (masked, no scaling) ----------------
// block = 1 wave. grid = (BT, 61 query tiles). Online softmax over 31 pair-tiles of 32 keys.
__global__ __launch_bounds__(32) void k_attn(const _Float16* __restrict__ Qh,
                                             const _Float16* __restrict__ Kh,
                                             const _Float16* __restrict__ Vt,
                                             const unsigned char* __restrict__ geo,
                                             float* __restrict__ sem) {
  __shared__ float sS[16][36];
  __shared__ float rM[16], rL[16], rSc[16];
  int bt = blockIdx.x, tq = blockIdx.y;
  int lane = threadIdx.x & 31, ln = lane & 15, kg = (lane >> 4) & 1, kg8 = kg * 8;
  int m0 = tq * 16;
  if (lane < 16) { rM[lane] = NEGBIG; rL[lane] = 0.f; }
  const _Float16* qrow = Qh + ((size_t)bt * MP + m0 + ln) * 64;
  v16h aq0 = a_frag_h(qrow,      lane);
  v16h aq1 = a_frag_h(qrow + 32, lane);
  int iRow[8], wRow[8];
#pragma unroll
  for (int r = 0; r < 8; ++r) {
    int mm = m0 + r + kg8;
    int wm = (mm < Mreal) ? (mm / N_) : 3;
    wRow[r] = wm;
    iRow[r] = (wm < 3) ? (mm - wm * N_) : 0;
  }
  v8f o0 = V8Z, o1 = V8Z, o2 = V8Z, o3 = V8Z;
  __syncthreads();

  for (int p = 0; p < 31; ++p) {
    int n0 = p * 32;
    if (p < 30) __builtin_prefetch(Kh + ((size_t)bt * MP + n0 + 32 + ln) * 64, 0, 1);
    const _Float16* krowA = Kh + ((size_t)bt * MP + n0 + ln) * 64;
    const _Float16* krowB = krowA + (size_t)16 * 64;
    v8f s0 = V8Z, s1 = V8Z;
    s0 = wmma16(aq0, b_frag_h(krowA,      lane), s0);
    s0 = wmma16(aq1, b_frag_h(krowA + 32, lane), s0);
    s1 = wmma16(aq0, b_frag_h(krowB,      lane), s1);
    s1 = wmma16(aq1, b_frag_h(krowB + 32, lane), s1);

    int na = n0 + ln, nb = na + 16;
    bool oa = (na >= Mreal), ob = (nb >= Mreal);
    int wa = oa ? 0 : na / N_; int ja = oa ? 0 : na - wa * N_;
    int wb = ob ? 0 : nb / N_; int jb = ob ? 0 : nb - wb * N_;
#pragma unroll
    for (int r = 0; r < 8; ++r) {
      int row = r + kg8;
      bool ma = oa || (wRow[r] >= 3);
      if (!ma) {
        int dw = wRow[r] - wa; dw = dw < 0 ? -dw : dw;
        ma = (dw == 2) || (dw == 1 ? (iRow[r] != ja) : (geo[iRow[r] * N_ + ja] != 0));
      }
      bool mb = ob || (wRow[r] >= 3);
      if (!mb) {
        int dw = wRow[r] - wb; dw = dw < 0 ? -dw : dw;
        mb = (dw == 2) || (dw == 1 ? (iRow[r] != jb) : (geo[iRow[r] * N_ + jb] != 0));
      }
      sS[row][ln]      = ma ? NEGBIG : s0[r];
      sS[row][16 + ln] = mb ? NEGBIG : s1[r];
    }
    __syncthreads();
    if (lane < 16) {
      float mx = NEGBIG;
#pragma unroll
      for (int c = 0; c < 32; ++c) mx = fmaxf(mx, sS[lane][c]);
      float mold = rM[lane];
      float mnew = fmaxf(mold, mx);
      float sc = __expf(mold - mnew);
      float sum = 0.f;
#pragma unroll
      for (int c = 0; c < 32; ++c) sum += __expf(sS[lane][c] - mnew);
      rL[lane] = rL[lane] * sc + sum;
      rM[lane] = mnew;
      rSc[lane] = sc;
    }
    __syncthreads();
    float scr[8];
#pragma unroll
    for (int r = 0; r < 8; ++r) scr[r] = rSc[r + kg8];
#pragma unroll
    for (int r = 0; r < 8; ++r) { o0[r] *= scr[r]; o1[r] *= scr[r]; o2[r] *= scr[r]; o3[r] *= scr[r]; }
    float rm = rM[ln];
    v16h ap;
#pragma unroll
    for (int i = 0; i < 8; ++i) {
      ap[i]     = (_Float16)__expf(sS[ln][kg8 + i] - rm);
      ap[i + 8] = (_Float16)__expf(sS[ln][16 + kg8 + i] - rm);
    }
    const _Float16* vbase = Vt + (size_t)bt * 64 * MP + n0;
    o0 = wmma16(ap, b_frag_h(vbase + (size_t)(0  + ln) * MP, lane), o0);
    o1 = wmma16(ap, b_frag_h(vbase + (size_t)(16 + ln) * MP, lane), o1);
    o2 = wmma16(ap, b_frag_h(vbase + (size_t)(32 + ln) * MP, lane), o2);
    o3 = wmma16(ap, b_frag_h(vbase + (size_t)(48 + ln) * MP, lane), o3);
    __syncthreads();
  }
  float li[8];
#pragma unroll
  for (int r = 0; r < 8; ++r) li[r] = 1.f / rL[r + kg8];
  v8f oo[4] = {o0, o1, o2, o3};
#pragma unroll
  for (int ot = 0; ot < 4; ++ot)
#pragma unroll
    for (int r = 0; r < 8; ++r) {
      int row = m0 + r + kg8;
      if (row < Mreal)
        sem[((size_t)bt * MP + row) * 64 + ot * 16 + ln] = oo[ot][r] * li[r];
    }
}

// ---------------- GCN pass 1: X1 = L * X0 (all 3 windows) ----------------
__global__ __launch_bounds__(32) void k_gcn1(const _Float16* __restrict__ gsoh,
                                             const _Float16* __restrict__ X0t,
                                             const float* __restrict__ x,
                                             const float* __restrict__ p12,
                                             const float* __restrict__ p21,
                                             const float* __restrict__ p23,
                                             const float* __restrict__ p32,
                                             _Float16* __restrict__ X1t,
                                             _Float16* __restrict__ X1r) {
  int bt = blockIdx.x;
  int w = blockIdx.y / 21, tile = blockIdx.y % 21;
  int b = bt / T_, t = bt % T_;
  int lane = threadIdx.x & 31, ln = lane & 15, kg = (lane >> 4) & 1;
  int i0 = tile * 16;
  const _Float16* arow = gsoh + ((size_t)b * NR + i0 + ln) * NP;
  v8f c[4] = {V8Z, V8Z, V8Z, V8Z};
  for (int kk = 0; kk < 11; ++kk) {
    v16h a = a_frag_h(arow + kk * 32, lane);
#pragma unroll
    for (int dt = 0; dt < 4; ++dt) {
      const _Float16* bcol = X0t + (((size_t)bt * 3 + w) * 64 + dt * 16 + ln) * NP + kk * 32;
      c[dt] = wmma16(a, b_frag_h(bcol, lane), c[dt]);
    }
  }
  int tm = (t - 1 < 0) ? 0 : t - 1;
  int tp = (t + 1 >= T_) ? T_ - 1 : t + 1;
#pragma unroll
  for (int dt = 0; dt < 4; ++dt)
#pragma unroll
    for (int r = 0; r < 8; ++r) {
      int i = i0 + r + kg * 8;
      int d = dt * 16 + ln;
      float v = c[dt][r];
      if (i < N_) {
        if (w == 0)      v += p21[i] * x[(((size_t)b * T_ + t)  * N_ + i) * 64 + d];
        else if (w == 1) v += p12[i] * x[(((size_t)b * T_ + tm) * N_ + i) * 64 + d]
                            + p32[i] * x[(((size_t)b * T_ + tp) * N_ + i) * 64 + d];
        else             v += p23[i] * x[(((size_t)b * T_ + t)  * N_ + i) * 64 + d];
      } else v = 0.f;
      X1t[(((size_t)bt * 3 + w) * 64 + d) * NP + i] = (_Float16)v;
      if (w == 1) X1r[((size_t)bt * NR + i) * 64 + d] = (_Float16)v;
    }
}

// ---------------- GCN pass 2: X2(center) = 2*(L*X1)(center) - X0(center) ----------------
__global__ __launch_bounds__(32) void k_gcn2(const _Float16* __restrict__ gsoh,
                                             const _Float16* __restrict__ X1t,
                                             const float* __restrict__ x,
                                             const float* __restrict__ p12,
                                             const float* __restrict__ p32,
                                             _Float16* __restrict__ X2r) {
  int bt = blockIdx.x, tile = blockIdx.y;
  int b = bt / T_, t = bt % T_;
  int lane = threadIdx.x & 31, ln = lane & 15, kg = (lane >> 4) & 1;
  int i0 = tile * 16;
  const _Float16* arow = gsoh + ((size_t)b * NR + i0 + ln) * NP;
  v8f c[4] = {V8Z, V8Z, V8Z, V8Z};
  for (int kk = 0; kk < 11; ++kk) {
    v16h a = a_frag_h(arow + kk * 32, lane);
#pragma unroll
    for (int dt = 0; dt < 4; ++dt) {
      const _Float16* bcol = X1t + (((size_t)bt * 3 + 1) * 64 + dt * 16 + ln) * NP + kk * 32;
      c[dt] = wmma16(a, b_frag_h(bcol, lane), c[dt]);
    }
  }
#pragma unroll
  for (int dt = 0; dt < 4; ++dt)
#pragma unroll
    for (int r = 0; r < 8; ++r) {
      int i = i0 + r + kg * 8;
      int d = dt * 16 + ln;
      float v = 0.f;
      if (i < N_) {
        float x0  = x[(((size_t)b * T_ + t) * N_ + i) * 64 + d];
        float x1a = (float)X1t[(((size_t)bt * 3 + 0) * 64 + d) * NP + i];
        float x1c = (float)X1t[(((size_t)bt * 3 + 2) * 64 + d) * NP + i];
        v = 2.f * (c[dt][r] + p12[i] * x1a + p32[i] * x1c) - x0;
      }
      X2r[((size_t)bt * NR + i) * 64 + d] = (_Float16)v;
    }
}

// ---------------- GCN pass 3: H = sum_k Xk * cheb_k (center) + LayerNorm ----------------
__global__ __launch_bounds__(32) void k_gcn3(const float* __restrict__ x,
                                             const _Float16* __restrict__ X1r,
                                             const _Float16* __restrict__ X2r,
                                             const _Float16* __restrict__ chebT,
                                             const float* __restrict__ ln_g,
                                             const float* __restrict__ ln_b,
                                             float* __restrict__ gcnO) {
  __shared__ float sH[16][68];
  int bt = blockIdx.x, tile = blockIdx.y;
  int b = bt / T_, t = bt % T_;
  int lane = threadIdx.x & 31, ln = lane & 15, kg = (lane >> 4) & 1;
  int i0 = tile * 16;
  int i = i0 + ln;
  bool valid = i < N_;
  const float* xrow = x + (((size_t)b * T_ + t) * N_ + (valid ? i : 0)) * 64;
  v16h a00 = a_frag_f(xrow,      lane, valid);
  v16h a01 = a_frag_f(xrow + 32, lane, valid);
  const _Float16* x1row = X1r + ((size_t)bt * NR + i0 + ln) * 64;
  v16h a10 = a_frag_h(x1row,      lane);
  v16h a11 = a_frag_h(x1row + 32, lane);
  const _Float16* x2row = X2r + ((size_t)bt * NR + i0 + ln) * 64;
  v16h a20 = a_frag_h(x2row,      lane);
  v16h a21 = a_frag_h(x2row + 32, lane);
  v8f c[4] = {V8Z, V8Z, V8Z, V8Z};
#pragma unroll
  for (int dt = 0; dt < 4; ++dt) {
    int j = dt * 16 + ln;
    const _Float16* c0 = chebT + ((size_t)(0 * 64 + j)) * 64;
    const _Float16* c1 = chebT + ((size_t)(1 * 64 + j)) * 64;
    const _Float16* c2 = chebT + ((size_t)(2 * 64 + j)) * 64;
    c[dt] = wmma16(a00, b_frag_h(c0,      lane), c[dt]);
    c[dt] = wmma16(a01, b_frag_h(c0 + 32, lane), c[dt]);
    c[dt] = wmma16(a10, b_frag_h(c1,      lane), c[dt]);
    c[dt] = wmma16(a11, b_frag_h(c1 + 32, lane), c[dt]);
    c[dt] = wmma16(a20, b_frag_h(c2,      lane), c[dt]);
    c[dt] = wmma16(a21, b_frag_h(c2 + 32, lane), c[dt]);
  }
#pragma unroll
  for (int dt = 0; dt < 4; ++dt)
#pragma unroll
    for (int r = 0; r < 8; ++r) sH[r + kg * 8][dt * 16 + ln] = c[dt][r];
  __syncthreads();
  if (lane < 16) {
    int ii = i0 + lane;
    if (ii < N_) {
      float mu = 0.f;
#pragma unroll
      for (int d = 0; d < 64; ++d) mu += sH[lane][d];
      mu *= (1.f / 64.f);
      float var = 0.f;
#pragma unroll
      for (int d = 0; d < 64; ++d) { float df = sH[lane][d] - mu; var += df * df; }
      var *= (1.f / 64.f);
      float inv = rsqrtf(var + 1e-5f);
      for (int d = 0; d < 64; ++d)
        gcnO[((size_t)bt * N_ + ii) * 64 + d] = (sH[lane][d] - mu) * inv * ln_g[d] + ln_b[d];
    }
  }
}

// ---------------- global mean of xa = xc + rc per (b, channel) ----------------
__global__ __launch_bounds__(256) void k_redmean(const float* __restrict__ sem,
                                                 const float* __restrict__ gcnO,
                                                 float* __restrict__ xmean) {
  __shared__ float red[256];
  int b = blockIdx.x, ch = blockIdx.y;
  float acc = 0.f;
  for (int idx = threadIdx.x; idx < T_ * N_; idx += 256) {
    int t = idx / N_, n = idx % N_;
    int bt = b * T_ + t;
    int flat = (N_ + n) * 64 + ch;
    int d2 = flat / Mreal, m2 = flat - d2 * Mreal;
    float xc = sem[((size_t)bt * MP + m2) * 64 + d2];
    float rc = gcnO[((size_t)bt * N_ + n) * 64 + ch];
    acc += xc + rc;
  }
  red[threadIdx.x] = acc;
  __syncthreads();
  for (int s = 128; s > 0; s >>= 1) {
    if (threadIdx.x < s) red[threadIdx.x] += red[threadIdx.x + s];
    __syncthreads();
  }
  if (threadIdx.x == 0) xmean[b * 64 + ch] = red[0] / (float)(T_ * N_);
}

// ---------------- global MLP branch ----------------
__global__ __launch_bounds__(64) void k_gmlp(const float* __restrict__ xmean,
                                             const float* __restrict__ gw1, const float* __restrict__ gb1,
                                             const float* __restrict__ gg1, const float* __restrict__ gbe1,
                                             const float* __restrict__ gw2, const float* __restrict__ gb2,
                                             const float* __restrict__ gg2, const float* __restrict__ gbe2,
                                             float* __restrict__ xg) {
  __shared__ float sm[64];
  __shared__ float sz[16];
  int b = blockIdx.x, tid = threadIdx.x;
  sm[tid] = xmean[b * 64 + tid];
  __syncthreads();
  if (tid < 16) {
    float a = gb1[tid];
    for (int d = 0; d < 64; ++d) a += gw1[tid * 64 + d] * sm[d];
    a = bn_eval(a, gg1[tid], gbe1[tid]);
    sz[tid] = fmaxf(a, 0.f);
  }
  __syncthreads();
  float a = gb2[tid];
  for (int i = 0; i < 16; ++i) a += gw2[tid * 16 + i] * sz[i];
  xg[b * 64 + tid] = bn_eval(a, gg2[tid], gbe2[tid]);
}

// ---------------- AFF fusion + output ----------------
__global__ __launch_bounds__(64) void k_fuse(const float* __restrict__ sem,
                                             const float* __restrict__ gcnO,
                                             const float* __restrict__ xg,
                                             const float* __restrict__ lw1, const float* __restrict__ lb1,
                                             const float* __restrict__ lg1, const float* __restrict__ lbe1,
                                             const float* __restrict__ lw2, const float* __restrict__ lb2,
                                             const float* __restrict__ lg2, const float* __restrict__ lbe2,
                                             float* __restrict__ out) {
  __shared__ float sa[64];
  __shared__ float sz[16];
  int pix = blockIdx.x;
  int n = pix % N_;
  int t = (pix / N_) % T_;
  int b = pix / (N_ * T_);
  int d = threadIdx.x;
  int bt = b * T_ + t;
  int flat = (N_ + n) * 64 + d;
  int d2 = flat / Mreal, m2 = flat - d2 * Mreal;
  float xc = sem[((size_t)bt * MP + m2) * 64 + d2];
  float rc = gcnO[((size_t)bt * N_ + n) * 64 + d];
  sa[d] = xc + rc;
  __syncthreads();
  if (d < 16) {
    float a = lb1[d];
    for (int k = 0; k < 64; ++k) a += lw1[d * 64 + k] * sa[k];
    a = bn_eval(a, lg1[d], lbe1[d]);
    sz[d] = fmaxf(a, 0.f);
  }
  __syncthreads();
  float a = lb2[d];
  for (int i = 0; i < 16; ++i) a += lw2[d * 16 + i] * sz[i];
  float xl = bn_eval(a, lg2[d], lbe2[d]);
  float wei = 1.f / (1.f + __expf(-(xl + xg[b * 64 + d])));
  out[((size_t)(bt)*N_ + n) * 64 + d] = 2.f * xc * wei + 2.f * rc * (1.f - wei);
}

// ---------------- host launcher ----------------
extern "C" void kernel_launch(void* const* d_in, const int* in_sizes, int n_in,
                              void* d_out, int out_size, void* d_ws, size_t ws_size,
                              hipStream_t stream) {
  (void)in_sizes; (void)n_in; (void)out_size; (void)ws_size;
  const float* x    = (const float*)d_in[0];
  const float* gso  = (const float*)d_in[1];
  const unsigned char* geo = (const unsigned char*)d_in[2];
  const float* Wq   = (const float*)d_in[3];
  const float* Wk   = (const float*)d_in[4];
  const float* Wv   = (const float*)d_in[5];
  const float* cheb = (const float*)d_in[6];
  const float* p12  = (const float*)d_in[7];
  const float* p21  = (const float*)d_in[8];
  const float* p23  = (const float*)d_in[9];
  const float* p32  = (const float*)d_in[10];
  const float* ln_g = (const float*)d_in[11];
  const float* ln_b = (const float*)d_in[12];
  const float* l_w1 = (const float*)d_in[13];
  const float* l_b1 = (const float*)d_in[14];
  const float* l_g1 = (const float*)d_in[15];
  const float* l_be1= (const float*)d_in[16];
  const float* l_w2 = (const float*)d_in[17];
  const float* l_b2 = (const float*)d_in[18];
  const float* l_g2 = (const float*)d_in[19];
  const float* l_be2= (const float*)d_in[20];
  const float* g_w1 = (const float*)d_in[21];
  const float* g_b1 = (const float*)d_in[22];
  const float* g_g1 = (const float*)d_in[23];
  const float* g_be1= (const float*)d_in[24];
  const float* g_w2 = (const float*)d_in[25];
  const float* g_b2 = (const float*)d_in[26];
  const float* g_g2 = (const float*)d_in[27];
  const float* g_be2= (const float*)d_in[28];

  char* ws = (char*)d_ws;
  size_t off = 0;
  auto take = [&](size_t bytes) { size_t o = off; off += (bytes + 255) & ~(size_t)255; return o; };
  size_t oQ    = take((size_t)BT_ * MP * 64 * 2);
  size_t oK    = take((size_t)BT_ * MP * 64 * 2);
  size_t oV    = take((size_t)BT_ * MP * 64 * 2);
  size_t oSem  = take((size_t)BT_ * MP * 64 * 4);
  size_t oX0t  = take((size_t)BT_ * 3 * 64 * NP * 2);
  size_t oX1t  = take((size_t)BT_ * 3 * 64 * NP * 2);
  size_t oX1r  = take((size_t)BT_ * NR * 64 * 2);
  size_t oX2r  = take((size_t)BT_ * NR * 64 * 2);
  size_t oG    = take((size_t)BT_ * N_ * 64 * 4);
  size_t oGso  = take((size_t)B_ * NR * NP * 2);
  size_t oCheb = take((size_t)3 * 64 * 64 * 2);
  size_t oXm   = take((size_t)B_ * 64 * 4);
  size_t oXg   = take((size_t)B_ * 64 * 4);

  _Float16* Qh    = (_Float16*)(ws + oQ);
  _Float16* Kh    = (_Float16*)(ws + oK);
  _Float16* Vt    = (_Float16*)(ws + oV);
  float*    sem   = (float*)(ws + oSem);
  _Float16* X0t   = (_Float16*)(ws + oX0t);
  _Float16* X1t   = (_Float16*)(ws + oX1t);
  _Float16* X1r   = (_Float16*)(ws + oX1r);
  _Float16* X2r   = (_Float16*)(ws + oX2r);
  float*    gcnO  = (float*)(ws + oG);
  _Float16* gsoh  = (_Float16*)(ws + oGso);
  _Float16* chebT = (_Float16*)(ws + oCheb);
  float*    xmean = (float*)(ws + oXm);
  float*    xgb   = (float*)(ws + oXg);
  float*    out   = (float*)d_out;

  {
    size_t n = (size_t)B_ * NR * NP;
    k_prep_gso<<<(unsigned)((n + 255) / 256), 256, 0, stream>>>(gso, gsoh);
  }
  k_prep_cheb<<<(3 * 64 * 64 + 255) / 256, 256, 0, stream>>>(cheb, chebT);
  {
    size_t n = (size_t)BT_ * 3 * 64 * NP;
    k_prep_x0t<<<(unsigned)((n + 255) / 256), 256, 0, stream>>>(x, X0t);
  }
  k_qkv <<<dim3(BT_, MP / 16), 96, 0, stream>>>(x, Wq, Wk, Wv, Qh, Kh, Vt);
  k_attn<<<dim3(BT_, 61),      32, 0, stream>>>(Qh, Kh, Vt, geo, sem);
  k_gcn1<<<dim3(BT_, 63),      32, 0, stream>>>(gsoh, X0t, x, p12, p21, p23, p32, X1t, X1r);
  k_gcn2<<<dim3(BT_, 21),      32, 0, stream>>>(gsoh, X1t, x, p12, p32, X2r);
  k_gcn3<<<dim3(BT_, 21),      32, 0, stream>>>(x, X1r, X2r, chebT, ln_g, ln_b, gcnO);
  k_redmean<<<dim3(B_, 64),   256, 0, stream>>>(sem, gcnO, xmean);
  k_gmlp<<<B_, 64, 0, stream>>>(xmean, g_w1, g_b1, g_g1, g_be1, g_w2, g_b2, g_g2, g_be2, xgb);
  k_fuse<<<B_ * T_ * N_, 64, 0, stream>>>(sem, gcnO, xgb,
                                          l_w1, l_b1, l_g1, l_be1, l_w2, l_b2, l_g2, l_be2, out);
}
